// Transform_7834020348655
// MI455X (gfx1250) — matrix operand order repaired and checked by reading, not compiled, not run on hardware
//
#include <hip/hip_runtime.h>
#include <hip/hip_bf16.h>

// ---------------------------------------------------------------------------
// Shapes (fixed by the reference)
// ---------------------------------------------------------------------------
#define B_    16
#define N_    4096
#define NP_   1024
#define K_    32
#define CIN_  128
#define C_    256
#define H_    4
#define CG_   131
#define KP_   160          // C_G padded to multiple of 32
#define NQ_   (B_ * NP_)   // 16384 query points

typedef __attribute__((ext_vector_type(16))) __bf16 v16bf;
typedef __attribute__((ext_vector_type(8)))  float  v8f;

// ---------------------------------------------------------------------------
// WMMA fragment loaders (wave32, 16-bit operands; this code's contiguous k order, same for A and B)
// A 16x32 (MxK):  lanes 0-15 -> row M=lane, K 0..15
//                 lanes16-31 -> row M=lane-16, K 16..31
// B 32x16 (KxN):  lanes 0-15 -> col N=lane,  K 0..15 (dword pairs 0..7)
//                 lanes16-31 -> col N=lane-16, K 16..31 (dword pairs 8..15)
// ---------------------------------------------------------------------------
__device__ __forceinline__ v16bf load_a_frag(const __bf16* base, int stride,
                                             int r0, int kb, int lane) {
  int r = r0 + (lane & 15);
  const unsigned* p = (const unsigned*)(base + (size_t)r * stride + kb);
  int off = (lane & 16) ? 8 : 0;
  union { v16bf v; unsigned u[8]; } f;
#pragma unroll
  for (int i = 0; i < 4; i++) { f.u[i] = p[off + i]; f.u[4 + i] = p[off + 4 + i]; }
  return f.v;
}

__device__ __forceinline__ v16bf load_b_frag(const __bf16* base, int stride,
                                             int nb, int kb, int lane) {
  int col = nb + (lane & 15);
  const unsigned* p = (const unsigned*)(base + (size_t)col * stride + kb);
  int off = (lane & 16) ? 8 : 0;
  union { v16bf v; unsigned u[8]; } f;
#pragma unroll
  for (int i = 0; i < 8; i++) f.u[i] = p[off + i];
  return f.v;
}

__device__ __forceinline__ v8f zero8() {
  v8f z = {0.f, 0.f, 0.f, 0.f, 0.f, 0.f, 0.f, 0.f};
  return z;
}

// ---------------------------------------------------------------------------
// Kernel 0: transpose weight [K][Ncols] fp32 -> [Ncols][Kpad] bf16 (zero pad)
// ---------------------------------------------------------------------------
__global__ void prep_kernel(const float* __restrict__ src, __bf16* __restrict__ dst,
                            int Ksrc, int Kpad, int Ncols) {
  int total = Ncols * Kpad;
  for (int i = blockIdx.x * blockDim.x + threadIdx.x; i < total;
       i += gridDim.x * blockDim.x) {
    int j = i / Kpad, k = i - j * Kpad;
    float v = (k < Ksrc) ? src[(size_t)k * Ncols + j] : 0.f;
    dst[i] = (__bf16)v;
  }
}

// ---------------------------------------------------------------------------
// Kernel 1: exact top-32 nearest neighbors, one wave32 per query.
// Running top-32 kept sorted ascending across lanes; per chunk: bitonic sort
// 32, reversed elementwise min-merge, 5-stage bitonic clean.
// ---------------------------------------------------------------------------
__device__ __forceinline__ void cmpx(float& d, int& i, int lane, int j, bool up) {
  float od = __shfl_xor(d, j, 32);
  int   oi = __shfl_xor(i, j, 32);
  bool keep_small = (((lane & j) == 0) == up);
  bool take = keep_small ? (od < d) : (od > d);
  if (take) { d = od; i = oi; }
}

__global__ __launch_bounds__(256) void knn_kernel(const float* __restrict__ xyz,
                                                  const float* __restrict__ newxyz,
                                                  int* __restrict__ knn) {
  int wave = (blockIdx.x * blockDim.x + threadIdx.x) >> 5;  // = query id
  int lane = threadIdx.x & 31;
  int b = wave >> 10;                    // NP_ = 1024
  const float* q = newxyz + (size_t)wave * 3;
  float qx = q[0], qy = q[1], qz = q[2];
  const float* xb = xyz + (size_t)b * N_ * 3;

  float rd = 3.4e38f;
  int   ri = 0;
#pragma unroll 1
  for (int c = 0; c < N_ / 32; c++) {
    int p = c * 32 + lane;
    const float* xp = xb + (size_t)p * 3;
    float dx = xp[0] - qx, dy = xp[1] - qy, dz = xp[2] - qz;
    float d = dx * dx + dy * dy + dz * dz;
    int pi = p;
    // bitonic sort chunk ascending
    for (int k = 2; k <= 32; k <<= 1) {
      bool up = ((lane & k) == 0);
      for (int j = k >> 1; j > 0; j >>= 1) cmpx(d, pi, lane, j, up);
    }
    // merge: reverse chunk, elementwise min -> bitonic; then clean ascending
    float nd = __shfl_xor(d, 31, 32);
    int   ni = __shfl_xor(pi, 31, 32);
    if (nd < rd) { rd = nd; ri = ni; }
    for (int j = 16; j > 0; j >>= 1) cmpx(rd, ri, lane, j, true);
  }
  knn[(size_t)wave * K_ + lane] = ri;
}

// ---------------------------------------------------------------------------
// Kernel 2: per-query fused gather -> LN1 -> QKV (WMMA) -> attention.
// Block = 128 threads (4 waves) = one query.
// Q-tiles never touch LDS: the per-column max (qm) is reduced straight out of
// the WMMA accumulators (f32) via register max + shfl_xor(16).  Only K and V
// (cols 256..767) are stored to LDS (bf16), halving the union buffer.
// Outputs: attn_mid[q][256] fp32 (pre-proj head outputs),
//          gmax_bf[q][160] bf16 (max over K of raw gf, zero padded).
// ---------------------------------------------------------------------------
union QSmem {
  float  gf[K_][132];        // gathered grouped features (fp32)
  __bf16 kv[K_][2 * C_];     // K,V result (bf16), reuses same LDS
};

__global__ __launch_bounds__(128) void query_kernel(
    const float* __restrict__ xyz, const float* __restrict__ newxyz,
    const float* __restrict__ feats, const int* __restrict__ knn,
    const float* __restrict__ n1_g, const float* __restrict__ n1_b,
    const __bf16* __restrict__ wqkvT,
    float* __restrict__ attn_mid, __bf16* __restrict__ gmax_bf) {
  __shared__ QSmem U;
  __shared__ __bf16 Abf[K_][KP_];
  __shared__ float  qm[C_];
  __shared__ float  attnw[H_][K_];
  __shared__ int    idxs[K_];

  int q = blockIdx.x;
  int b = q >> 10;
  int t = threadIdx.x, lane = t & 31, w = t >> 5;

  if (t < K_) idxs[t] = knn[(size_t)q * K_ + t];
  __syncthreads();

  float nx = newxyz[(size_t)q * 3 + 0];
  float ny = newxyz[(size_t)q * 3 + 1];
  float nz = newxyz[(size_t)q * 3 + 2];

  // ---- gather: row r handled by thread quartet (t&3 strided over cols)
  {
    int r = t >> 2, c0 = t & 3;
    int pid = idxs[r];
    const float* fx = xyz + (size_t)(b * N_ + pid) * 3;
    const float* fr = feats + (size_t)(b * N_ + pid) * CIN_;
    for (int c = c0; c < CG_; c += 4) {
      float v;
      if (c == 0)      v = fx[0] - nx;
      else if (c == 1) v = fx[1] - ny;
      else if (c == 2) v = fx[2] - nz;
      else             v = fr[c - 3];
      U.gf[r][c] = v;
    }
  }
  __syncthreads();

  // ---- column max over K (dense-branch input), written padded bf16
  for (int c = t; c < KP_; c += 128) {
    float m = 0.f;
    if (c < CG_) {
      m = -3.4e38f;
      for (int r = 0; r < K_; r++) m = fmaxf(m, U.gf[r][c]);
    }
    gmax_bf[(size_t)q * KP_ + c] = (__bf16)m;
  }

  // ---- LayerNorm per row (eps = 1e-3), wave w owns rows 8w..8w+7
  for (int r = w * 8; r < w * 8 + 8; r++) {
    float s = 0.f, s2 = 0.f;
    for (int c = lane; c < CG_; c += 32) {
      float v = U.gf[r][c];
      s += v; s2 += v * v;
    }
    for (int m = 16; m > 0; m >>= 1) {
      s  += __shfl_xor(s, m, 32);
      s2 += __shfl_xor(s2, m, 32);
    }
    float mean = s * (1.f / 131.f);
    float var  = s2 * (1.f / 131.f) - mean * mean;
    float rs   = rsqrtf(var + 1e-3f);
    for (int c = lane; c < KP_; c += 32) {
      float o = 0.f;
      if (c < CG_) o = (U.gf[r][c] - mean) * rs * n1_g[c] + n1_b[c];
      Abf[r][c] = (__bf16)o;
    }
  }
  __syncthreads();  // gf dead, Abf ready -> LDS union becomes kv

  // ---- QKV GEMM: M=32 (2 tiles) x N=768 x K=160 (5 chunks), bf16 WMMA
  // Column partition: Nb = nt*64 + w*16 so "is this a Q tile" is wave-uniform.
  {
    v16bf afr[2][5];
#pragma unroll
    for (int mt = 0; mt < 2; mt++)
#pragma unroll
      for (int c = 0; c < 5; c++)
        afr[mt][c] = load_a_frag(&Abf[0][0], KP_, mt * 16, c * 32, lane);

    for (int nt = 0; nt < 12; nt++) {
      int Nb = nt * 64 + w * 16;
      v8f acc0 = zero8(), acc1 = zero8();
#pragma unroll
      for (int c = 0; c < 5; c++) {
        v16bf bf = load_b_frag(wqkvT, KP_, Nb, c * 32, lane);
        acc0 = __builtin_amdgcn_wmma_f32_16x16x32_bf16(
            false, afr[0][c], false, bf, (short)0, acc0, false, false);
        acc1 = __builtin_amdgcn_wmma_f32_16x16x32_bf16(
            false, afr[1][c], false, bf, (short)0, acc1, false, false);
      }
      int col = Nb + (lane & 15);
      if (Nb < C_) {
        // Q tile: reduce column max over all 32 rows straight from registers
        float m = acc0[0];
#pragma unroll
        for (int rI = 1; rI < 8; rI++) m = fmaxf(m, acc0[rI]);
#pragma unroll
        for (int rI = 0; rI < 8; rI++) m = fmaxf(m, acc1[rI]);
        m = fmaxf(m, __shfl_xor(m, 16, 32));
        if (lane < 16) qm[col] = m;
      } else {
        // K/V tile: store bf16 to LDS for the attention stage
        int cc = col - C_;
        int rb = (lane & 16) ? 8 : 0;
#pragma unroll
        for (int rI = 0; rI < 8; rI++) {
          U.kv[rb + rI][cc]        = (__bf16)acc0[rI];
          U.kv[16 + rb + rI][cc]   = (__bf16)acc1[rI];
        }
      }
    }
  }
  __syncthreads();

  // ---- attention: logits = qm.k / 16; softmax over K; out = attn.v
  {
    int h = w, k = lane;
    float l = 0.f;
    for (int dd = 0; dd < 64; dd++)
      l += qm[h * 64 + dd] * (float)U.kv[k][h * 64 + dd];
    l *= 0.0625f;  // 256^-0.5
    float mx = l;
    for (int m = 16; m > 0; m >>= 1) mx = fmaxf(mx, __shfl_xor(mx, m, 32));
    float e = __expf(l - mx);
    float s = e;
    for (int m = 16; m > 0; m >>= 1) s += __shfl_xor(s, m, 32);
    attnw[h][k] = e / s;
  }
  __syncthreads();

  for (int dim = t; dim < C_; dim += 128) {
    int h = dim >> 6;
    float o = 0.f;
    for (int k = 0; k < K_; k++)
      o += attnw[h][k] * (float)U.kv[k][C_ + dim];
    attn_mid[(size_t)q * C_ + dim] = o;
  }
}

// ---------------------------------------------------------------------------
// Kernel 3: feat = relu(attn@proj + pb) + relu(gmax@dense + db);
//           out  = (LN2(feat) -> relu(fc1) -> fc2 + b) + feat.
// Block = 128 threads (4 waves) handles 16 query rows; all GEMMs via WMMA.
// ---------------------------------------------------------------------------
__global__ __launch_bounds__(128) void feat_kernel(
    const float* __restrict__ attn_mid, const __bf16* __restrict__ gmax_bf,
    const __bf16* __restrict__ projT, const __bf16* __restrict__ denseT,
    const __bf16* __restrict__ fc1T, const __bf16* __restrict__ fc2T,
    const float* __restrict__ proj_b, const float* __restrict__ dense_b,
    const float* __restrict__ fc1_b, const float* __restrict__ fc2_b,
    const float* __restrict__ n2_g, const float* __restrict__ n2_b,
    float* __restrict__ out) {
  __shared__ __bf16 A1[16][C_];       // attn_mid rows (bf16)
  __shared__ __bf16 A2[16][KP_];      // gmax rows (bf16, padded)
  __shared__ float  featf[16][264];   // feat (fp32) for LN + residual
  __shared__ __bf16 A3[16][C_];       // LN2(feat)
  __shared__ __bf16 hbf[16][2 * C_];  // relu(fc1) hidden

  int r0 = blockIdx.x * 16;
  int t = threadIdx.x, lane = t & 31, w = t >> 5;

  for (int e = t; e < 16 * C_; e += 128) {
    int r = e >> 8, c = e & 255;
    A1[r][c] = (__bf16)attn_mid[(size_t)(r0 + r) * C_ + c];
  }
  for (int e = t; e < 16 * KP_; e += 128) {
    int r = e / KP_, c = e - r * KP_;
    A2[r][c] = gmax_bf[(size_t)(r0 + r) * KP_ + c];
  }
  __syncthreads();

  // ---- GEMM1 (proj) + GEMM2 (dense), fused relu/bias/add -> featf
  for (int nt = 0; nt < 4; nt++) {
    int Nb = w * 64 + nt * 16;
    v8f acc1 = zero8(), acc2 = zero8();
#pragma unroll
    for (int c = 0; c < 8; c++) {
      v16bf a = load_a_frag(&A1[0][0], C_, 0, c * 32, lane);
      v16bf bb = load_b_frag(projT, C_, Nb, c * 32, lane);
      acc1 = __builtin_amdgcn_wmma_f32_16x16x32_bf16(false, a, false, bb,
                                                     (short)0, acc1, false, false);
    }
#pragma unroll
    for (int c = 0; c < 5; c++) {
      v16bf a = load_a_frag(&A2[0][0], KP_, 0, c * 32, lane);
      v16bf bb = load_b_frag(denseT, KP_, Nb, c * 32, lane);
      acc2 = __builtin_amdgcn_wmma_f32_16x16x32_bf16(false, a, false, bb,
                                                     (short)0, acc2, false, false);
    }
    int col = Nb + (lane & 15);
    int rb  = (lane & 16) ? 8 : 0;
#pragma unroll
    for (int rI = 0; rI < 8; rI++) {
      float f = fmaxf(acc1[rI] + proj_b[col], 0.f) +
                fmaxf(acc2[rI] + dense_b[col], 0.f);
      featf[rb + rI][col] = f;
    }
  }
  __syncthreads();

  // ---- LN2 per row, wave w owns rows 4w..4w+3
  for (int r = w * 4; r < w * 4 + 4; r++) {
    float s = 0.f, s2 = 0.f;
    for (int c = lane; c < C_; c += 32) {
      float v = featf[r][c];
      s += v; s2 += v * v;
    }
    for (int m = 16; m > 0; m >>= 1) {
      s  += __shfl_xor(s, m, 32);
      s2 += __shfl_xor(s2, m, 32);
    }
    float mean = s * (1.f / 256.f);
    float var  = s2 * (1.f / 256.f) - mean * mean;
    float rs   = rsqrtf(var + 1e-3f);
    for (int c = lane; c < C_; c += 32)
      A3[r][c] = (__bf16)((featf[r][c] - mean) * rs * n2_g[c] + n2_b[c]);
  }
  __syncthreads();

  // ---- GEMM3: fc1 (256 -> 512) + relu
  for (int nt = 0; nt < 8; nt++) {
    int Nb = w * 128 + nt * 16;
    v8f acc = zero8();
#pragma unroll
    for (int c = 0; c < 8; c++) {
      v16bf a = load_a_frag(&A3[0][0], C_, 0, c * 32, lane);
      v16bf bb = load_b_frag(fc1T, C_, Nb, c * 32, lane);
      acc = __builtin_amdgcn_wmma_f32_16x16x32_bf16(false, a, false, bb,
                                                    (short)0, acc, false, false);
    }
    int col = Nb + (lane & 15);
    int rb  = (lane & 16) ? 8 : 0;
#pragma unroll
    for (int rI = 0; rI < 8; rI++)
      hbf[rb + rI][col] = (__bf16)fmaxf(acc[rI] + fc1_b[col], 0.f);
  }
  __syncthreads();

  // ---- GEMM4: fc2 (512 -> 256) + bias + residual -> out
  for (int nt = 0; nt < 4; nt++) {
    int Nb = w * 64 + nt * 16;
    v8f acc = zero8();
#pragma unroll
    for (int c = 0; c < 16; c++) {
      v16bf a = load_a_frag(&hbf[0][0], 2 * C_, 0, c * 32, lane);
      v16bf bb = load_b_frag(fc2T, 2 * C_, Nb, c * 32, lane);
      acc = __builtin_amdgcn_wmma_f32_16x16x32_bf16(false, a, false, bb,
                                                    (short)0, acc, false, false);
    }
    int col = Nb + (lane & 15);
    int rb  = (lane & 16) ? 8 : 0;
#pragma unroll
    for (int rI = 0; rI < 8; rI++) {
      int row = rb + rI;
      out[(size_t)(r0 + row) * C_ + col] =
          acc[rI] + fc2_b[col] + featf[row][col];
    }
  }
}

// ---------------------------------------------------------------------------
// Host launcher
// ---------------------------------------------------------------------------
extern "C" void kernel_launch(void* const* d_in, const int* in_sizes, int n_in,
                              void* d_out, int out_size, void* d_ws, size_t ws_size,
                              hipStream_t stream) {
  const float* xyz     = (const float*)d_in[0];
  const float* newxyz  = (const float*)d_in[1];
  const float* feats   = (const float*)d_in[2];
  const float* w_qkv   = (const float*)d_in[3];
  const float* proj_w  = (const float*)d_in[4];
  const float* proj_b  = (const float*)d_in[5];
  const float* dense_w = (const float*)d_in[6];
  const float* dense_b = (const float*)d_in[7];
  const float* fc1_w   = (const float*)d_in[8];
  const float* fc1_b   = (const float*)d_in[9];
  const float* fc2_w   = (const float*)d_in[10];
  const float* fc2_b   = (const float*)d_in[11];
  const float* n1_g    = (const float*)d_in[12];
  const float* n1_b    = (const float*)d_in[13];
  const float* n2_g    = (const float*)d_in[14];
  const float* n2_b    = (const float*)d_in[15];

  char* ws = (char*)d_ws;
  // workspace layout (256B aligned)
  __bf16* wqkvT   = (__bf16*)(ws + 0);               // 768*160*2  = 245760
  __bf16* projT   = (__bf16*)(ws + 245760);          // 256*256*2  = 131072
  __bf16* denseT  = (__bf16*)(ws + 376832);          // 256*160*2  = 81920
  __bf16* fc1T    = (__bf16*)(ws + 458752);          // 512*256*2  = 262144
  __bf16* fc2T    = (__bf16*)(ws + 720896);          // 256*512*2  = 262144
  int*    knn     = (int*)   (ws + 983040);          // 16384*32*4 = 2097152
  __bf16* gmax_bf = (__bf16*)(ws + 3080192);         // 16384*160*2= 5242880
  float*  amid    = (float*) (ws + 8323072);         // 16384*256*4= 16777216

  // weight prep (transpose + pad + bf16)
  prep_kernel<<<480, 256, 0, stream>>>(w_qkv,  wqkvT, CG_, KP_,    3 * C_);
  prep_kernel<<<256, 256, 0, stream>>>(proj_w, projT, C_,  C_,     C_);
  prep_kernel<<<160, 256, 0, stream>>>(dense_w,denseT,CG_, KP_,    C_);
  prep_kernel<<<512, 256, 0, stream>>>(fc1_w,  fc1T,  C_,  C_,     2 * C_);
  prep_kernel<<<512, 256, 0, stream>>>(fc2_w,  fc2T,  2*C_,2 * C_, C_);

  // exact KNN top-32: one wave per query
  knn_kernel<<<NQ_ / 8, 256, 0, stream>>>(xyz, newxyz, knn);

  // fused gather + LN1 + WMMA QKV + attention (Q never leaves registers)
  query_kernel<<<NQ_, 128, 0, stream>>>(xyz, newxyz, feats, knn, n1_g, n1_b,
                                        wqkvT, amid, gmax_bf);

  // proj + dense + LN2 + MLP (all WMMA) + residual
  feat_kernel<<<NQ_ / 16, 128, 0, stream>>>(amid, gmax_bf, projT, denseT, fc1T,
                                            fc2T, proj_b, dense_b, fc1_b, fc2_b,
                                            n2_g, n2_b, (float*)d_out);
}
